// SpectralLayerWithAttention_13743895347615
// MI455X (gfx1250) — matrix operand order repaired and checked by reading, not compiled, not run on hardware
//
#include <hip/hip_runtime.h>
#include <hip/hip_bf16.h>
#include <math.h>

#ifndef __has_builtin
#define __has_builtin(x) 0
#endif

#if __has_builtin(__builtin_amdgcn_global_load_async_to_lds_b128)
#define HAVE_ASYNC_LDS 1
#else
#define HAVE_ASYNC_LDS 0
#endif

// ---------------------------------------------------------------------------
// Types for gfx1250 WMMA (wave32): v_wmma_f32_16x16x32_bf16
// ---------------------------------------------------------------------------
typedef __attribute__((ext_vector_type(8)))  __bf16 v8bf;
typedef __attribute__((ext_vector_type(16))) __bf16 v16bf;
typedef __attribute__((ext_vector_type(8)))  float  v8f;
typedef __attribute__((ext_vector_type(4)))  int    v4i;

typedef __attribute__((address_space(1))) v4i* gv4i_ptr;   // global int32x4*
typedef __attribute__((address_space(3))) v4i* lv4i_ptr;   // LDS    int32x4*

__device__ __forceinline__ unsigned short f2bf(float f) {
  unsigned int u = __float_as_uint(f);
  unsigned int r = u + 0x7FFFu + ((u >> 16) & 1u);   // round-to-nearest-even
  return (unsigned short)(r >> 16);
}

__device__ __forceinline__ void wait_async_all() {
#if HAVE_ASYNC_LDS
# if __has_builtin(__builtin_amdgcn_s_wait_asynccnt)
  __builtin_amdgcn_s_wait_asynccnt(0);
# else
  asm volatile("s_wait_asynccnt 0" ::: "memory");
# endif
#endif
}

// ---------------------------------------------------------------------------
// LDS tile geometry: 128 rows x 32 K of bf16, row stride padded to 40 elements
// (80 B = 20 dwords) so consecutive rows spread across LDS banks and every
// 16-byte chunk stays 16 B aligned.
// ---------------------------------------------------------------------------
#define TILE_ROWS   128
#define TILE_K      32
#define ROW_PITCH   40          // bf16 elements (64 B payload + 16 B pad)

// Stage one 128x32 bf16 slab (row-major, leading dim = ld elements) into LDS.
// 256 threads: each covers 2 rows' worth of one 16 B chunk (4 chunks/row).
__device__ __forceinline__ void stage_tile(const __bf16* __restrict__ src, int ld,
                                           int row0, int k0, __bf16* lds) {
  const int t = threadIdx.x;        // 0..255
  const int c = t & 3;              // 16B chunk within the 64B row
  const int r = t >> 2;             // 0..63
#pragma unroll
  for (int p = 0; p < 2; ++p) {
    const int row = r + p * 64;
    const __bf16* gp = src + (size_t)(row0 + row) * ld + k0 + c * 8;
    __bf16* lp = lds + row * ROW_PITCH + c * 8;
#if HAVE_ASYNC_LDS
    void* gv = (void*)const_cast<__bf16*>(gp);
    void* lv = (void*)lp;
    __builtin_amdgcn_global_load_async_to_lds_b128(
        (gv4i_ptr)gv, (lv4i_ptr)lv, 0, 0);
#else
    v8bf v = *(const v8bf*)gp;
    *(v8bf*)lp = v;
#endif
  }
}

// ---------------------------------------------------------------------------
// A-fragment (16x32 bf16) from an LDS tile:
//   lanes 0-15 : M = lane,    chunks at K 0..7  and K 16..23
//   lanes 16-31: M = lane-16, chunks at K 8..15 and K 24..31
// ---------------------------------------------------------------------------
__device__ __forceinline__ v16bf lds_a_frag(const __bf16* tile, int row0, int lane) {
  const int r  = row0 + (lane & 15);
  const int hi = lane >> 4;
  const __bf16* p = tile + r * ROW_PITCH + hi * 8;
  v8bf lo = *(const v8bf*)p;
  v8bf hv = *(const v8bf*)(p + 16);
  v16bf out;
#pragma unroll
  for (int i = 0; i < 8; ++i) { out[i] = lo[i]; out[8 + i] = hv[i]; }
  return out;
}

// B-fragment (32x16 bf16) from an LDS tile holding Bt rows (N-major):
//   lanes 0-15 : col = lane,    chunks at K 0..7  and K 16..23
//   lanes 16-31: col = lane-16, chunks at K 8..15 and K 24..31
__device__ __forceinline__ v16bf lds_b_frag(const __bf16* tile, int col0, int lane) {
  const int c  = col0 + (lane & 15);
  const int hi = lane >> 4;
  const __bf16* p = tile + c * ROW_PITCH + hi * 8;
  v8bf lo = *(const v8bf*)p;
  v8bf hv = *(const v8bf*)(p + 16);
  v16bf out;
#pragma unroll
  for (int i = 0; i < 8; ++i) { out[i] = lo[i]; out[8 + i] = hv[i]; }
  return out;
}

// ---------------------------------------------------------------------------
// bf16 WMMA GEMM: C(M,N) = A(M,K) * Bt(N,K)^T  with fused epilogues.
//   mode 0: Out.f32 = acc
//   mode 1: Out.f32 = res + alpha*acc
//   mode 2: Out.bf16 = gelu(acc + bias)            (exact erf GELU)
//   mode 3: Out.f32 = res + alpha*(acc + bias)
// Block: 256 threads = 8 waves arranged 4(M) x 2(N); wave tile 32x64;
// block tile 128x128; K-step 32; A/B slabs double-buffered in LDS via
// global_load_async_to_lds_b128 (ASYNCcnt) when available.
// ---------------------------------------------------------------------------
__global__ __launch_bounds__(256)
void gemm_bf16_wmma(const __bf16* __restrict__ A, const __bf16* __restrict__ Bt,
                    void* __restrict__ Out, const float* __restrict__ bias,
                    const float* __restrict__ res,
                    int M, int N, int K, float alpha, int mode) {
  __shared__ __bf16 sA[2][TILE_ROWS * ROW_PITCH];
  __shared__ __bf16 sB[2][TILE_ROWS * ROW_PITCH];

  const int lane = threadIdx.x & 31;
  const int wave = threadIdx.x >> 5;
  const int wm   = wave >> 1;          // 0..3  (M direction)
  const int wn   = wave & 1;           // 0..1  (N direction)
  const int blkRow = blockIdx.y * 128;
  const int blkCol = blockIdx.x * 128;

  v8f acc[2][4];
#pragma unroll
  for (int i = 0; i < 2; ++i)
#pragma unroll
    for (int j = 0; j < 4; ++j)
#pragma unroll
      for (int e = 0; e < 8; ++e) acc[i][j][e] = 0.0f;

  // Prologue: stage K-slab 0 into buffer 0.
  stage_tile(A,  K, blkRow, 0, sA[0]);
  stage_tile(Bt, K, blkCol, 0, sB[0]);

  int buf = 0;
  for (int k0 = 0; k0 < K; k0 += 32, buf ^= 1) {
    wait_async_all();        // our slab of buf is in LDS
    __syncthreads();         // everyone's slab is in LDS

    if (k0 + 32 < K) {       // overlap next slab's copy with this slab's math
      stage_tile(A,  K, blkRow, k0 + 32, sA[buf ^ 1]);
      stage_tile(Bt, K, blkCol, k0 + 32, sB[buf ^ 1]);
    }

    v16bf a0 = lds_a_frag(sA[buf], wm * 32,      lane);
    v16bf a1 = lds_a_frag(sA[buf], wm * 32 + 16, lane);
    v16bf b0 = lds_b_frag(sB[buf], wn * 64,      lane);
    v16bf b1 = lds_b_frag(sB[buf], wn * 64 + 16, lane);
    v16bf b2 = lds_b_frag(sB[buf], wn * 64 + 32, lane);
    v16bf b3 = lds_b_frag(sB[buf], wn * 64 + 48, lane);

    acc[0][0] = __builtin_amdgcn_wmma_f32_16x16x32_bf16(false, a0, false, b0, (short)0, acc[0][0], false, false);
    acc[0][1] = __builtin_amdgcn_wmma_f32_16x16x32_bf16(false, a0, false, b1, (short)0, acc[0][1], false, false);
    acc[0][2] = __builtin_amdgcn_wmma_f32_16x16x32_bf16(false, a0, false, b2, (short)0, acc[0][2], false, false);
    acc[0][3] = __builtin_amdgcn_wmma_f32_16x16x32_bf16(false, a0, false, b3, (short)0, acc[0][3], false, false);
    acc[1][0] = __builtin_amdgcn_wmma_f32_16x16x32_bf16(false, a1, false, b0, (short)0, acc[1][0], false, false);
    acc[1][1] = __builtin_amdgcn_wmma_f32_16x16x32_bf16(false, a1, false, b1, (short)0, acc[1][1], false, false);
    acc[1][2] = __builtin_amdgcn_wmma_f32_16x16x32_bf16(false, a1, false, b2, (short)0, acc[1][2], false, false);
    acc[1][3] = __builtin_amdgcn_wmma_f32_16x16x32_bf16(false, a1, false, b3, (short)0, acc[1][3], false, false);

    __syncthreads();         // all reads of buf done before it is restaged
  }

  // Epilogue. C/D layout: VGPR r -> row ((lane>=16)?8:0)+r, col = lane&15.
  const int col   = lane & 15;
  const int rbase = (lane >> 4) * 8;
#pragma unroll
  for (int i = 0; i < 2; ++i) {
#pragma unroll
    for (int j = 0; j < 4; ++j) {
      const int row0 = blkRow + wm * 32 + i * 16 + rbase;
      const int c0   = blkCol + wn * 64 + j * 16 + col;
#pragma unroll
      for (int r = 0; r < 8; ++r) {
        const size_t idx = (size_t)(row0 + r) * N + c0;
        float v = acc[i][j][r];
        if (mode == 0) {
          ((float*)Out)[idx] = v;
        } else if (mode == 1) {
          ((float*)Out)[idx] = res[idx] + alpha * v;
        } else if (mode == 2) {
          float t = v + bias[c0];
          float g = 0.5f * t * (1.0f + erff(t * 0.70710678118654752f));
          ((unsigned short*)Out)[idx] = f2bf(g);
        } else {
          ((float*)Out)[idx] = res[idx] + alpha * (v + bias[c0]);
        }
      }
    }
  }
}

// ---------------------------------------------------------------------------
// LayerNorm over D, emit bf16. One 256-thread block (8 waves) per row.
// ---------------------------------------------------------------------------
__global__ __launch_bounds__(256)
void ln_to_bf16(const float* __restrict__ x, const float* __restrict__ g,
                const float* __restrict__ b, unsigned short* __restrict__ out, int D) {
  __shared__ float s_sum[8], s_sq[8];
  const size_t row = blockIdx.x;
  const float* xr = x + row * (size_t)D;

  float lsum = 0.0f, lsq = 0.0f;
  for (int i = threadIdx.x; i < D; i += blockDim.x) {
    float v = xr[i]; lsum += v; lsq += v * v;
  }
#pragma unroll
  for (int off = 16; off > 0; off >>= 1) {
    lsum += __shfl_xor(lsum, off, 32);
    lsq  += __shfl_xor(lsq,  off, 32);
  }
  const int wave = threadIdx.x >> 5;
  if ((threadIdx.x & 31) == 0) { s_sum[wave] = lsum; s_sq[wave] = lsq; }
  __syncthreads();
  if (wave == 0) {
    lsum = (threadIdx.x < 8) ? s_sum[threadIdx.x] : 0.0f;
    lsq  = (threadIdx.x < 8) ? s_sq[threadIdx.x]  : 0.0f;
#pragma unroll
    for (int off = 4; off > 0; off >>= 1) {
      lsum += __shfl_xor(lsum, off, 32);
      lsq  += __shfl_xor(lsq,  off, 32);
    }
    if (threadIdx.x == 0) { s_sum[0] = lsum; s_sq[0] = lsq; }
  }
  __syncthreads();
  const float invD = 1.0f / (float)D;
  const float mean = s_sum[0] * invD;
  const float var  = s_sq[0] * invD - mean * mean;
  const float rstd = rsqrtf(var + 1e-5f);
  unsigned short* orow = out + row * (size_t)D;
  for (int i = threadIdx.x; i < D; i += blockDim.x) {
    orow[i] = f2bf((xr[i] - mean) * rstd * g[i] + b[i]);
  }
}

// ---------------------------------------------------------------------------
// Sequential spectral scan: one thread per (b, f). Coalesced across f.
// Emits bf16 directly (A-matrix of GEMM2).
// ---------------------------------------------------------------------------
__global__ __launch_bounds__(256)
void spectral_scan(const float* __restrict__ spec, const float* __restrict__ log_decay,
                   const float* __restrict__ freq, unsigned short* __restrict__ so,
                   int Bc, int S, int F) {
  const int tid = blockIdx.x * blockDim.x + threadIdx.x;
  if (tid >= Bc * F) return;
  const int b = tid / F, f = tid % F;
  const float decay = 1.0f / (1.0f + expf(-log_decay[f]));
  const float omega = tanhf(freq[f]) * 0.1f;
  const float c = cosf(omega), s = sinf(omega);
  float sr = 0.0f, si = 0.0f;
  const size_t base = (size_t)b * S * (2 * F) + f;
  for (int t = 0; t < S; ++t) {
    const size_t o = base + (size_t)t * (2 * F);
    float r  = spec[o];
    float im = spec[o + F];
    sr = sr * decay + r  * 0.1f;
    si = si * decay + im * 0.1f;
    float nr = fminf(10.0f, fmaxf(-10.0f, sr * c - si * s));
    float ni = fminf(10.0f, fmaxf(-10.0f, sr * s + si * c));
    sr = nr; si = ni;
    so[o]     = f2bf(nr);
    so[o + F] = f2bf(ni);
  }
}

// ---------------------------------------------------------------------------
// Elementwise f32 -> bf16 (weight conversion).
// ---------------------------------------------------------------------------
__global__ __launch_bounds__(256)
void cvt_f32_bf16(const float* __restrict__ in, unsigned short* __restrict__ out, int n) {
  const int i = blockIdx.x * blockDim.x + threadIdx.x;
  if (i < n) out[i] = f2bf(in[i]);
}

// ---------------------------------------------------------------------------
extern "C" void kernel_launch(void* const* d_in, const int* in_sizes, int n_in,
                              void* d_out, int out_size, void* d_ws, size_t ws_size,
                              hipStream_t stream) {
  (void)in_sizes; (void)n_in; (void)out_size; (void)ws_size;

  const float* x         = (const float*)d_in[0];
  const float* W_to      = (const float*)d_in[1];
  const float* log_decay = (const float*)d_in[2];
  const float* freqs     = (const float*)d_in[3];
  const float* W_from    = (const float*)d_in[4];
  const float* ln1_g     = (const float*)d_in[5];
  const float* ln1_b     = (const float*)d_in[6];
  const float* ln3_g     = (const float*)d_in[7];
  const float* ln3_b     = (const float*)d_in[8];
  const float* W_ffn1    = (const float*)d_in[9];
  const float* b_ffn1    = (const float*)d_in[10];
  const float* W_ffn2    = (const float*)d_in[11];
  const float* b_ffn2    = (const float*)d_in[12];

  constexpr int Bc = 4, S = 4096, D = 1024, F = 256;
  constexpr int M  = Bc * S;   // 16384
  constexpr int F2 = 2 * F;    // 512
  constexpr int E  = 4 * D;    // 4096

  // Workspace carve-out (all 256B aligned).
  char* w = (char*)d_ws;
  auto carve = [&](size_t bytes) -> char* {
    char* p = w; w += (bytes + 255) & ~(size_t)255; return p;
  };
  unsigned short* hbf   = (unsigned short*)carve((size_t)M * D * 2);   // LN1 out, bf16
  float*          spec  = (float*)carve((size_t)M * F2 * 4);           // GEMM1 out
  unsigned short* sobf  = (unsigned short*)carve((size_t)M * F2 * 2);  // scan out, bf16
  float*          x2    = (float*)carve((size_t)M * D * 4);            // residual 2
  unsigned short* x2bf  = (unsigned short*)carve((size_t)M * D * 2);   // LN3 out, bf16
  unsigned short* h3bf  = (unsigned short*)carve((size_t)M * E * 2);   // FFN mid, bf16
  unsigned short* wto   = (unsigned short*)carve((size_t)F2 * D * 2);
  unsigned short* wfrom = (unsigned short*)carve((size_t)D * F2 * 2);
  unsigned short* wf1   = (unsigned short*)carve((size_t)E * D * 2);
  unsigned short* wf2   = (unsigned short*)carve((size_t)D * E * 2);

  // Weight conversions f32 -> bf16.
  cvt_f32_bf16<<<dim3((F2 * D + 255) / 256), 256, 0, stream>>>(W_to,   wto,   F2 * D);
  cvt_f32_bf16<<<dim3((D * F2 + 255) / 256), 256, 0, stream>>>(W_from, wfrom, D * F2);
  cvt_f32_bf16<<<dim3((E * D + 255) / 256), 256, 0, stream>>>(W_ffn1, wf1,   E * D);
  cvt_f32_bf16<<<dim3((D * E + 255) / 256), 256, 0, stream>>>(W_ffn2, wf2,   D * E);

  // 1) LN1 -> bf16
  ln_to_bf16<<<dim3(M), 256, 0, stream>>>(x, ln1_g, ln1_b, hbf, D);

  // 2) spec = h @ W_to^T   (M x F2, K = D)
  gemm_bf16_wmma<<<dim3(F2 / 128, M / 128), 256, 0, stream>>>(
      (const __bf16*)hbf, (const __bf16*)wto, spec, nullptr, nullptr,
      M, F2, D, 0.0f, 0);

  // 3) sequential spectral scan -> bf16
  spectral_scan<<<dim3((Bc * F + 255) / 256), 256, 0, stream>>>(
      spec, log_decay, freqs, sobf, Bc, S, F);

  // 4) x2 = x + 0.1 * (so @ W_from^T)   (M x D, K = F2)
  gemm_bf16_wmma<<<dim3(D / 128, M / 128), 256, 0, stream>>>(
      (const __bf16*)sobf, (const __bf16*)wfrom, x2, nullptr, x,
      M, D, F2, 0.1f, 1);

  // 5) LN3 -> bf16
  ln_to_bf16<<<dim3(M), 256, 0, stream>>>(x2, ln3_g, ln3_b, x2bf, D);

  // 6) h3 = gelu(x2ln @ W_ffn1^T + b1) -> bf16   (M x E, K = D)
  gemm_bf16_wmma<<<dim3(E / 128, M / 128), 256, 0, stream>>>(
      (const __bf16*)x2bf, (const __bf16*)wf1, h3bf, b_ffn1, nullptr,
      M, E, D, 0.0f, 2);

  // 7) out = x2 + 0.1 * (h3 @ W_ffn2^T + b2)   (M x D, K = E)
  gemm_bf16_wmma<<<dim3(D / 128, M / 128), 256, 0, stream>>>(
      (const __bf16*)h3bf, (const __bf16*)wf2, d_out, b_ffn2, x2,
      M, D, E, 0.1f, 3);
}
